// Decoder_78159814853289
// MI455X (gfx1250) — matrix operand order repaired and checked by reading, not compiled, not run on hardware
//
#include <hip/hip_runtime.h>
#include <hip/hip_bf16.h>
#include <math.h>
#include <type_traits>

// ---------------- types ----------------
typedef _Float16 half_t;
typedef __attribute__((ext_vector_type(16))) _Float16 v16h;
typedef __attribute__((ext_vector_type(8)))  _Float16 v8h;
typedef __attribute__((ext_vector_type(8)))  float    v8f;
typedef __attribute__((ext_vector_type(4)))  float    v4f;

#define COV_DIM 4
#define EMBED   64
#define E_DIM   256
#define LH_DIM  512
#define NH      8
#define DH      32
#define PS      48
#define START   336
#define B_SIZE  512
#define T_LEN   336
#define K1_PAD  864   // [x(5) | embed(64) | ctx(256) | 0(27) | h1(512)]
#define K2_PAD  768   // [h1(512) | h2(256)]

// =====================================================================
// Generic WMMA GEMM:  C[M,N] = A[M,K] * W[N,K]^T + bias
// A: f32 or f16 row-major (lda, 16-byte aligned rows at 32-elem K chunks),
// W: f16 row-major N x K. Workgroup tile 128(M) x 64(N); 8 waves,
// each wave 32x32 via 4 v_wmma_f32_16x16x32_f16 per K-chunk.
// K % 32 == 0, M % 128 == 0, N % 64 == 0.
// =====================================================================
template <typename AT>
__global__ __launch_bounds__(256)
void wmma_gemm_kernel(const AT* __restrict__ A, int lda,
                      const half_t* __restrict__ W, int ldw,
                      const float* __restrict__ bias,
                      float* __restrict__ Cf, int ldc,
                      half_t* __restrict__ Ch, int ldch,
                      int K)
{
    const int n0  = blockIdx.x * 64;
    const int m0  = blockIdx.y * 128;
    const int tid = threadIdx.x;
    const int lane = tid & 31;
    const int wave = tid >> 5;
    const int wm = wave & 3;   // 0..3 -> +32*wm rows
    const int wn = wave >> 2;  // 0..1 -> +32*wn cols
    const int g  = lane >> 4;  // lane half-group
    const int ln = lane & 15;

    __shared__ half_t As[128][40];  // 80B row stride (16B aligned)
    __shared__ half_t Bs[64][40];

    const int a_row = tid >> 1;
    const int a_kb  = (tid & 1) * 16;
    const int b_row = tid >> 2;
    const int b_kb  = (tid & 3) * 8;
    const AT*     a_src = A + (size_t)(m0 + a_row) * lda + a_kb;
    const half_t* b_src = W + (size_t)(n0 + b_row) * ldw + b_kb;

    v8f acc[2][2];
#pragma unroll
    for (int i = 0; i < 2; i++)
#pragma unroll
        for (int j = 0; j < 2; j++)
#pragma unroll
            for (int e = 0; e < 8; e++) acc[i][j][e] = 0.0f;

    for (int k0 = 0; k0 < K; k0 += 32) {
        __syncthreads();
        // ---- stage A tile (128 x 32) via 128-bit loads ----
        {
            half_t* dst = &As[a_row][a_kb];
            if constexpr (std::is_same<AT, _Float16>::value) {
                const v8h* s = (const v8h*)(a_src + k0);
                *(v8h*)(dst)     = s[0];
                *(v8h*)(dst + 8) = s[1];
            } else {
                const v4f* s = (const v4f*)(a_src + k0);
#pragma unroll
                for (int q = 0; q < 4; q++) {
                    v4f t = s[q];
#pragma unroll
                    for (int e = 0; e < 4; e++) dst[q * 4 + e] = (half_t)t[e];
                }
            }
        }
        // ---- stage B tile (64 n x 32 k) via 128-bit load ----
        *(v8h*)&Bs[b_row][b_kb] = *(const v8h*)(b_src + k0);

        // prefetch next K-chunk (global_prefetch_b8 on gfx1250)
        if (k0 + 32 < K) {
            __builtin_prefetch(a_src + k0 + 32, 0, 1);
            __builtin_prefetch(b_src + k0 + 32, 0, 1);
        }
        __syncthreads();

        // ---- build fragments per ISA 7.12.2 layouts ----
        v16h afrag[2], bfrag[2];
#pragma unroll
        for (int i = 0; i < 2; i++) {
            int row = wm * 32 + i * 16 + ln;
#pragma unroll
            for (int e = 0; e < 16; e++) {
                // A 16x32 f16: lanes 0-15: K=0..7,16..23; lanes 16-31: K=8..15,24..31
                int k = (e < 8) ? (g * 8 + e) : (16 + g * 8 + (e - 8));
                afrag[i][e] = As[row][k];
            }
        }
#pragma unroll
        for (int j = 0; j < 2; j++) {
            int col = wn * 32 + j * 16 + ln;
#pragma unroll
            for (int e = 0; e < 16; e++) {
                // B 32x16 f16: lanes 0-15: K=0..7,16..23; lanes 16-31: K=8..15,24..31
                bfrag[j][e] = Bs[col][(e < 8) ? (g * 8 + e) : (16 + g * 8 + (e - 8))];
            }
        }
#pragma unroll
        for (int i = 0; i < 2; i++)
#pragma unroll
            for (int j = 0; j < 2; j++)
                acc[i][j] = __builtin_amdgcn_wmma_f32_16x16x32_f16(
                    false, afrag[i], false, bfrag[j],
                    (short)0, acc[i][j], false, false);
    }

    // ---- epilogue: D VGPR v -> (M = base + g*8 + v, N = ln) ----
#pragma unroll
    for (int i = 0; i < 2; i++) {
        const int rowb = m0 + wm * 32 + i * 16 + g * 8;
#pragma unroll
        for (int j = 0; j < 2; j++) {
            const int col = n0 + wn * 32 + j * 16 + ln;
            const float bv = bias[col];
            float vals[8];
#pragma unroll
            for (int v = 0; v < 8; v++) vals[v] = acc[i][j][v] + bv;
            if (Cf) {
                float* p = Cf + (size_t)rowb * ldc + col;
#pragma unroll
                for (int v = 0; v < 8; v++) p[(size_t)v * ldc] = vals[v];
            }
            if (Ch) {
                half_t* p = Ch + (size_t)rowb * ldch + col;
#pragma unroll
                for (int v = 0; v < 8; v++) p[(size_t)v * ldch] = (half_t)vals[v];
            }
        }
    }
}

// =====================================================================
// Prep / glue kernels
// =====================================================================
__global__ void zero_f32_kernel(float* p, int n) {
    int i = blockIdx.x * blockDim.x + threadIdx.x;
    if (i < n) p[i] = 0.0f;
}
__global__ void zero_f16_kernel(half_t* p, int n) {
    int i = blockIdx.x * blockDim.x + threadIdx.x;
    if (i < n) p[i] = (half_t)0.0f;
}
__global__ void cvt_f32_f16_kernel(const float* __restrict__ s, half_t* __restrict__ d, int n) {
    int i = blockIdx.x * blockDim.x + threadIdx.x;
    if (i < n) d[i] = (half_t)s[i];
}
__global__ void add_vec_kernel(const float* a, const float* b, float* o, int n) {
    int i = blockIdx.x * blockDim.x + threadIdx.x;
    if (i < n) o[i] = a[i] + b[i];
}

// W1comb[2048][864]: 0..324 = w_ih1, 325..351 = 0, 352..863 = w_hh1
__global__ void prep_w1_kernel(const float* __restrict__ w_ih1, const float* __restrict__ w_hh1,
                               half_t* __restrict__ W) {
    int i = blockIdx.x * blockDim.x + threadIdx.x;
    if (i >= 4 * LH_DIM * K1_PAD) return;
    int row = i / K1_PAD, col = i % K1_PAD;
    float v;
    if (col < 325)      v = w_ih1[row * 325 + col];
    else if (col < 352) v = 0.0f;
    else                v = w_hh1[row * 512 + (col - 352)];
    W[i] = (half_t)v;
}
// W2comb[1024][768]: 0..511 = w_ih2, 512..767 = w_hh2
__global__ void prep_w2_kernel(const float* __restrict__ w_ih2, const float* __restrict__ w_hh2,
                               half_t* __restrict__ W) {
    int i = blockIdx.x * blockDim.x + threadIdx.x;
    if (i >= 4 * E_DIM * K2_PAD) return;
    int row = i / K2_PAD, col = i % K2_PAD;
    float v = (col < 512) ? w_ih2[row * 512 + col] : w_hh2[row * 256 + (col - 512)];
    W[i] = (half_t)v;
}

// constant act1 columns: embed (5..68); padding region already zeroed
__global__ void fill_embed_kernel(const float* __restrict__ idx_embed, half_t* __restrict__ act1) {
    int i = blockIdx.x * blockDim.x + threadIdx.x;
    if (i >= B_SIZE * EMBED) return;
    int b = i / EMBED, j = i % EMBED;
    act1[(size_t)b * K1_PAD + 5 + j] = (half_t)idx_embed[i];
}

// per-step x columns (0..4)
__global__ void set_x_kernel(const float* __restrict__ zxs, half_t* __restrict__ act1, int s) {
    int i = blockIdx.x * blockDim.x + threadIdx.x;
    if (i >= B_SIZE * (1 + COV_DIM)) return;
    int b = i / (1 + COV_DIM), j = i % (1 + COV_DIM);
    act1[(size_t)b * K1_PAD + j] =
        (half_t)zxs[((size_t)b * (START + PS) + (START + s)) * (1 + COV_DIM) + j];
}

__device__ __forceinline__ float sigmoidf(float x) { return 1.0f / (1.0f + __expf(-x)); }

// gates[b][4H] (i,f,g,o) -> c update; h written as optional f32 and up to two
// f16 destinations (pre-offset base pointers with given row strides).
__global__ void lstm_act_kernel(const float* __restrict__ gates, float* __restrict__ c,
                                float* __restrict__ hf,
                                half_t* __restrict__ hd0, int ld0,
                                half_t* __restrict__ hd1, int ld1, int H) {
    int idx = blockIdx.x * blockDim.x + threadIdx.x;
    if (idx >= B_SIZE * H) return;
    int b = idx / H, j = idx % H;
    const float* g = gates + (size_t)b * 4 * H;
    float ig = sigmoidf(g[j]);
    float fg = sigmoidf(g[H + j]);
    float gg = tanhf(g[2 * H + j]);
    float og = sigmoidf(g[3 * H + j]);
    float cn = fg * c[idx] + ig * gg;
    c[idx] = cn;
    float hn = og * tanhf(cn);
    if (hf)  hf[idx] = hn;
    if (hd0) hd0[(size_t)b * ld0 + j] = (half_t)hn;
    if (hd1) hd1[(size_t)b * ld1 + j] = (half_t)hn;
}

// Single-query cross attention per (head, batch). kp/vp: (B*T, 256) f16, col = h*32+d
__global__ __launch_bounds__(128)
void attention_kernel(const float* __restrict__ q,
                      const half_t* __restrict__ kp, const half_t* __restrict__ vp,
                      float* __restrict__ ctxh, float* __restrict__ attn0) {
    int h = blockIdx.x, b = blockIdx.y;
    __shared__ float sc[T_LEN];
    __shared__ float red[128];
    __shared__ float qs[DH];
    int tid = threadIdx.x;
    if (tid < DH) qs[tid] = q[(size_t)b * E_DIM + h * DH + tid];
    __syncthreads();
    const float scale = 0.1767766952966369f; // 1/sqrt(32)
    for (int t = tid; t < T_LEN; t += 128) {
        const half_t* kr = kp + ((size_t)b * T_LEN + t) * E_DIM + h * DH;
        float acc = 0.0f;
#pragma unroll
        for (int d = 0; d < DH; d++) acc += qs[d] * (float)kr[d];
        sc[t] = acc * scale;
    }
    __syncthreads();
    float m = -1e30f;
    for (int t = tid; t < T_LEN; t += 128) m = fmaxf(m, sc[t]);
    red[tid] = m; __syncthreads();
    for (int st = 64; st > 0; st >>= 1) {
        if (tid < st) red[tid] = fmaxf(red[tid], red[tid + st]);
        __syncthreads();
    }
    m = red[0]; __syncthreads();
    float sum = 0.0f;
    for (int t = tid; t < T_LEN; t += 128) { float e = __expf(sc[t] - m); sc[t] = e; sum += e; }
    red[tid] = sum; __syncthreads();
    for (int st = 64; st > 0; st >>= 1) {
        if (tid < st) red[tid] += red[tid + st];
        __syncthreads();
    }
    float inv = 1.0f / red[0];
    for (int t = tid; t < T_LEN; t += 128) sc[t] *= inv;
    __syncthreads();
    if (b == 0) {
        for (int t = tid; t < T_LEN; t += 128) attn0[h * T_LEN + t] = sc[t];
    }
    if (tid < DH) {
        float acc = 0.0f;
        for (int t = 0; t < T_LEN; t++)
            acc += sc[t] * (float)vp[((size_t)b * T_LEN + t) * E_DIM + h * DH + tid];
        ctxh[(size_t)b * E_DIM + h * DH + tid] = acc;
    }
}

// blocks 0..1: mu/sigma heads for b = blk*256+tid; block 2: head-avg attention row
__global__ void finish_step_kernel(const float* __restrict__ h2, const float* __restrict__ ctx,
                                   const float* __restrict__ attn0,
                                   const float* __restrict__ mu_w, const float* __restrict__ mu_b,
                                   const float* __restrict__ ps_w, const float* __restrict__ ps_b,
                                   float* __restrict__ out_mu, float* __restrict__ out_sigma,
                                   float* __restrict__ out_attn, int s) {
    int tid = threadIdx.x;
    if (blockIdx.x == 2) {
        for (int t = tid; t < T_LEN; t += 256) {
            float a = 0.0f;
#pragma unroll
            for (int h = 0; h < NH; h++) a += attn0[h * T_LEN + t];
            out_attn[t * PS + s] = a * (1.0f / NH);
        }
        return;
    }
    int b = blockIdx.x * 256 + tid;
    const float* h2r = h2 + (size_t)b * E_DIM;
    const float* cr  = ctx + (size_t)b * E_DIM;
    float am = 0.0f, as = 0.0f;
    for (int j = 0; j < E_DIM; j++) { am += mu_w[j] * h2r[j];        as += ps_w[j] * h2r[j]; }
    for (int j = 0; j < E_DIM; j++) { am += mu_w[E_DIM + j] * cr[j]; as += ps_w[E_DIM + j] * cr[j]; }
    am += mu_b[0];
    as += ps_b[0];
    float sp = (as > 20.0f) ? as : log1pf(__expf(as));
    out_mu[b * PS + s]    = am;
    out_sigma[b * PS + s] = sp;
}

// =====================================================================
// Launch
// =====================================================================
extern "C" void kernel_launch(void* const* d_in, const int* in_sizes, int n_in,
                              void* d_out, int out_size, void* d_ws, size_t ws_size,
                              hipStream_t stream) {
    const float* zxs       = (const float*)d_in[0];
    const float* key       = (const float*)d_in[1];
    const float* value     = (const float*)d_in[2];
    const float* idx_embed = (const float*)d_in[3];
    const float* w_ih1     = (const float*)d_in[4];
    const float* w_hh1     = (const float*)d_in[5];
    const float* b_ih1     = (const float*)d_in[6];
    const float* b_hh1     = (const float*)d_in[7];
    const float* w_ih2     = (const float*)d_in[8];
    const float* w_hh2     = (const float*)d_in[9];
    const float* b_ih2     = (const float*)d_in[10];
    const float* b_hh2     = (const float*)d_in[11];
    const float* in_proj_w = (const float*)d_in[12];
    const float* in_proj_b = (const float*)d_in[13];
    const float* out_proj_w= (const float*)d_in[14];
    const float* out_proj_b= (const float*)d_in[15];
    const float* mu_w      = (const float*)d_in[16];
    const float* mu_b      = (const float*)d_in[17];
    const float* ps_w      = (const float*)d_in[18];
    const float* ps_b      = (const float*)d_in[19];

    float* out_mu    = (float*)d_out;            // B*PS
    float* out_sigma = out_mu + B_SIZE * PS;     // B*PS
    float* out_attn  = out_sigma + B_SIZE * PS;  // T*PS

    // ---- workspace bump allocator ----
    char* wsp = (char*)d_ws;
    auto alloc = [&](size_t bytes) -> char* {
        char* p = wsp;
        wsp += (bytes + 255) & ~(size_t)255;
        return p;
    };
    const size_t BT = (size_t)B_SIZE * T_LEN;  // 172032
    half_t* W1c    = (half_t*)alloc((size_t)4 * LH_DIM * K1_PAD * 2);
    half_t* W2c    = (half_t*)alloc((size_t)4 * E_DIM * K2_PAD * 2);
    half_t* wq_h   = (half_t*)alloc((size_t)E_DIM * E_DIM * 2);
    half_t* wk_h   = (half_t*)alloc((size_t)E_DIM * E_DIM * 2);
    half_t* wv_h   = (half_t*)alloc((size_t)E_DIM * E_DIM * 2);
    half_t* wo_h   = (half_t*)alloc((size_t)E_DIM * E_DIM * 2);
    float*  bias1  = (float*)alloc((size_t)4 * LH_DIM * 4);
    float*  bias2  = (float*)alloc((size_t)4 * E_DIM * 4);
    half_t* kp_h   = (half_t*)alloc(BT * E_DIM * 2);   // 88 MB
    half_t* vp_h   = (half_t*)alloc(BT * E_DIM * 2);   // 88 MB
    half_t* act1   = (half_t*)alloc((size_t)B_SIZE * K1_PAD * 2);
    half_t* act2   = (half_t*)alloc((size_t)B_SIZE * K2_PAD * 2);
    float*  gates1 = (float*)alloc((size_t)B_SIZE * 4 * LH_DIM * 4);
    float*  gates2 = (float*)alloc((size_t)B_SIZE * 4 * E_DIM * 4);
    float*  c1     = (float*)alloc((size_t)B_SIZE * LH_DIM * 4);
    float*  h2     = (float*)alloc((size_t)B_SIZE * E_DIM * 4);
    float*  c2     = (float*)alloc((size_t)B_SIZE * E_DIM * 4);
    float*  ctx    = (float*)alloc((size_t)B_SIZE * E_DIM * 4);
    float*  qbuf   = (float*)alloc((size_t)B_SIZE * E_DIM * 4);
    float*  ctxh   = (float*)alloc((size_t)B_SIZE * E_DIM * 4);
    float*  attn0  = (float*)alloc((size_t)NH * T_LEN * 4);

    const float* bq = in_proj_b;
    const float* bk = in_proj_b + E_DIM;
    const float* bv = in_proj_b + 2 * E_DIM;

    // ---- per-call prep ----
    {
        int n = 4 * LH_DIM * K1_PAD;
        prep_w1_kernel<<<(n + 255) / 256, 256, 0, stream>>>(w_ih1, w_hh1, W1c);
        n = 4 * E_DIM * K2_PAD;
        prep_w2_kernel<<<(n + 255) / 256, 256, 0, stream>>>(w_ih2, w_hh2, W2c);
        n = E_DIM * E_DIM;
        cvt_f32_f16_kernel<<<(n + 255) / 256, 256, 0, stream>>>(in_proj_w, wq_h, n);
        cvt_f32_f16_kernel<<<(n + 255) / 256, 256, 0, stream>>>(in_proj_w + n, wk_h, n);
        cvt_f32_f16_kernel<<<(n + 255) / 256, 256, 0, stream>>>(in_proj_w + 2 * n, wv_h, n);
        cvt_f32_f16_kernel<<<(n + 255) / 256, 256, 0, stream>>>(out_proj_w, wo_h, n);
        add_vec_kernel<<<(4 * LH_DIM + 255) / 256, 256, 0, stream>>>(b_ih1, b_hh1, bias1, 4 * LH_DIM);
        add_vec_kernel<<<(4 * E_DIM + 255) / 256, 256, 0, stream>>>(b_ih2, b_hh2, bias2, 4 * E_DIM);
        // zero act buffers (covers ctx/h regions and K-padding for step 0)
        n = B_SIZE * K1_PAD;
        zero_f16_kernel<<<(n + 255) / 256, 256, 0, stream>>>(act1, n);
        n = B_SIZE * K2_PAD;
        zero_f16_kernel<<<(n + 255) / 256, 256, 0, stream>>>(act2, n);
        n = B_SIZE * EMBED;
        fill_embed_kernel<<<(n + 255) / 256, 256, 0, stream>>>(idx_embed, act1);
        zero_f32_kernel<<<(B_SIZE * LH_DIM + 255) / 256, 256, 0, stream>>>(c1, B_SIZE * LH_DIM);
        zero_f32_kernel<<<(B_SIZE * E_DIM + 255) / 256, 256, 0, stream>>>(h2, B_SIZE * E_DIM);
        zero_f32_kernel<<<(B_SIZE * E_DIM + 255) / 256, 256, 0, stream>>>(c2, B_SIZE * E_DIM);
        zero_f32_kernel<<<(B_SIZE * E_DIM + 255) / 256, 256, 0, stream>>>(ctx, B_SIZE * E_DIM);
    }

    // ---- hoisted K/V projections (memory-bound, WMMA, f16 out) ----
    {
        dim3 grid(E_DIM / 64, (unsigned)(BT / 128));
        wmma_gemm_kernel<float><<<grid, 256, 0, stream>>>(
            key, E_DIM, wk_h, E_DIM, bk, (float*)nullptr, 0, kp_h, E_DIM, E_DIM);
        wmma_gemm_kernel<float><<<grid, 256, 0, stream>>>(
            value, E_DIM, wv_h, E_DIM, bv, (float*)nullptr, 0, vp_h, E_DIM, E_DIM);
    }

    // ---- 48 sequential decode steps ----
    for (int s = 0; s < PS; s++) {
        int nx = B_SIZE * (1 + COV_DIM);
        set_x_kernel<<<(nx + 255) / 256, 256, 0, stream>>>(zxs, act1, s);

        // gates1 = act1 @ W1c^T + bias1   (512 x 2048 x 864)
        wmma_gemm_kernel<half_t><<<dim3(4 * LH_DIM / 64, B_SIZE / 128), 256, 0, stream>>>(
            act1, K1_PAD, W1c, K1_PAD, bias1, gates1, 4 * LH_DIM, (half_t*)nullptr, 0, K1_PAD);
        // h1 -> act1[352..863] and act2[0..511]
        lstm_act_kernel<<<(B_SIZE * LH_DIM + 255) / 256, 256, 0, stream>>>(
            gates1, c1, (float*)nullptr, act1 + 352, K1_PAD, act2, K2_PAD, LH_DIM);

        // gates2 = act2 @ W2c^T + bias2   (512 x 1024 x 768)
        wmma_gemm_kernel<half_t><<<dim3(4 * E_DIM / 64, B_SIZE / 128), 256, 0, stream>>>(
            act2, K2_PAD, W2c, K2_PAD, bias2, gates2, 4 * E_DIM, (half_t*)nullptr, 0, K2_PAD);
        // h2 -> f32 (heads) and act2[512..767] (q-proj input, next-step recurrence)
        lstm_act_kernel<<<(B_SIZE * E_DIM + 255) / 256, 256, 0, stream>>>(
            gates2, c2, h2, act2 + 512, K2_PAD, (half_t*)nullptr, 0, E_DIM);

        // q = h2 @ wq^T + bq  (f16 A read straight out of act2 tail)
        wmma_gemm_kernel<half_t><<<dim3(E_DIM / 64, B_SIZE / 128), 256, 0, stream>>>(
            act2 + 512, K2_PAD, wq_h, E_DIM, bq, qbuf, E_DIM, (half_t*)nullptr, 0, E_DIM);

        attention_kernel<<<dim3(NH, B_SIZE), 128, 0, stream>>>(qbuf, kp_h, vp_h, ctxh, attn0);

        // ctx = ctxh @ out_proj_w^T + b ; f16 copy lands in act1 ctx columns
        wmma_gemm_kernel<float><<<dim3(E_DIM / 64, B_SIZE / 128), 256, 0, stream>>>(
            ctxh, E_DIM, wo_h, E_DIM, out_proj_b, ctx, E_DIM, act1 + 69, K1_PAD, E_DIM);

        finish_step_kernel<<<3, 256, 0, stream>>>(
            h2, ctx, attn0, mu_w, mu_b, ps_w, ps_b, out_mu, out_sigma, out_attn, s);
    }
}
